// NonLocalBlockOriginal_34531537060181
// MI455X (gfx1250) — matrix operand order repaired and checked by reading, not compiled, not run on hardware
//
#include <hip/hip_runtime.h>
#include <hip/hip_bf16.h>

typedef __bf16 bf16;
typedef __attribute__((ext_vector_type(16))) __bf16 bf16x16;
typedef __attribute__((ext_vector_type(8)))  __bf16 bf16x8;
typedef __attribute__((ext_vector_type(8)))  float  f32x8;

#define Bn 4
#define Cc 128
#define CI 64
#define Hh 96
#define Ww 96
#define HW (Hh*Ww)       // 9216
#define Hp 48
#define Wp 48
#define Mp (Hp*Wp)       // 2304 pooled pixels
#define GN_GROUPS 32
#define GN_EPS 1e-5f
#define SM_SHIFT 20.0f   // fixed softmax shift (cancels in num/denom)

// ---------------- WMMA fragment helpers ----------------

__device__ __forceinline__ bf16x16 frag_cat(bf16x8 lo, bf16x8 hi) {
  bf16x16 v;
#pragma unroll
  for (int i = 0; i < 8; ++i) { v[i] = lo[i]; v[i + 8] = hi[i]; }
  return v;
}

// A-matrix fragment (16x32, MxK), row-major source with given row stride.
// Lane l<16: M=l, K = {kb+h*8 .. +7} U {kb+16+h*8 .. +7} with h = l>>4.
__device__ __forceinline__ bf16x16 load_a(const bf16* base, int r, int h,
                                          int stride, int kb) {
  const bf16* p = base + (size_t)r * stride + kb + h * 8;
  bf16x8 lo = *(const bf16x8*)p;
  bf16x8 hi = *(const bf16x8*)(p + 16);
  return frag_cat(lo, hi);
}

// B-matrix fragment (32x16, KxN). Source stores N rows of K values
// (i.e. element B[k][n] = base[n*stride + k]).
// Lane l<16: N=l, K = {kb+h*8 .. +7} U {kb+16+h*8 .. +7} with h = l>>4.
__device__ __forceinline__ bf16x16 load_b(const bf16* base, int r, int h,
                                          int stride, int kb) {
  const bf16* p = base + (size_t)r * stride + kb + h * 8;
  bf16x8 lo = *(const bf16x8*)p;
  bf16x8 hi = *(const bf16x8*)(p + 16);
  return frag_cat(lo, hi);
}

__device__ __forceinline__ f32x8 wmma_bf16(bf16x16 a, bf16x16 b, f32x8 c) {
  return __builtin_amdgcn_wmma_f32_16x16x32_bf16(false, a, false, b,
                                                 (short)0, c, false, false);
}

// ---------------- Kernel 1: theta = conv1x1(x, w_theta) -> bf16 [B][N][64] ----------------

__global__ __launch_bounds__(256) void k_theta(const float* __restrict__ x,
                                               const float* __restrict__ w,
                                               const float* __restrict__ bias,
                                               bf16* __restrict__ theta) {
  __shared__ float ws[CI * Cc];   // 32 KB weights
  __shared__ float xs[Cc * 64];   // 32 KB x tile [128 ch][64 pix]
  const int b = blockIdx.y;
  const int n0 = blockIdx.x * 64;
  const int tid = threadIdx.x;
  const float* xb = x + (size_t)b * Cc * HW;
  for (int i = tid; i < CI * Cc; i += 256) ws[i] = w[i];
  for (int i = tid; i < Cc * 64; i += 256) {
    int c = i >> 6, p = i & 63;
    xs[i] = xb[(size_t)c * HW + n0 + p];
  }
  __syncthreads();
  const int pix = tid & 63, og = tid >> 6;   // og uniform per wave pair
  float acc[16];
#pragma unroll
  for (int i = 0; i < 16; ++i) acc[i] = 0.f;
  for (int c = 0; c < Cc; ++c) {
    float xv = xs[c * 64 + pix];
#pragma unroll
    for (int i = 0; i < 16; ++i) acc[i] += ws[(og * 16 + i) * Cc + c] * xv;
  }
  bf16* dst = theta + ((size_t)b * HW + n0 + pix) * CI + og * 16;
  bf16x8 v0, v1;
#pragma unroll
  for (int i = 0; i < 8; ++i) {
    v0[i] = (bf16)(acc[i] + bias[og * 16 + i]);
    v1[i] = (bf16)(acc[8 + i] + bias[og * 16 + 8 + i]);
  }
  *(bf16x8*)dst = v0;
  *(bf16x8*)(dst + 8) = v1;
}

// ---------------- Kernel 2: phi/g projections + 2x2 maxpool ----------------
// phi -> bf16 [B][Mp][64] (row-major pixel x channel)
// gT  -> bf16 [B][64][Mp] (channel-major, for B-fragment loads)

__global__ __launch_bounds__(256) void k_phig(const float* __restrict__ x,
                                              const float* __restrict__ wphi,
                                              const float* __restrict__ bphi,
                                              const float* __restrict__ wg,
                                              const float* __restrict__ bg,
                                              bf16* __restrict__ phi,
                                              bf16* __restrict__ gT) {
  __shared__ float wsp[CI * Cc];
  __shared__ float wsg[CI * Cc];
  const int b = blockIdx.y;
  const int m0 = blockIdx.x * 64;
  const int tid = threadIdx.x;
  for (int i = tid; i < CI * Cc; i += 256) { wsp[i] = wphi[i]; wsg[i] = wg[i]; }
  __syncthreads();
  const int pix = tid & 63, og = tid >> 6;
  const int m = m0 + pix;
  const int hp = m / Wp, wq = m % Wp;
  const float* xb = x + (size_t)b * Cc * HW + (size_t)(2 * hp) * Ww + 2 * wq;
  float accP[16], accG[16];
#pragma unroll
  for (int i = 0; i < 16; ++i) { accP[i] = -3.0e38f; accG[i] = -3.0e38f; }
#pragma unroll 1
  for (int sp = 0; sp < 4; ++sp) {
    const float* xp = xb + (sp & 1) + (sp >> 1) * Ww;
    float tp[16], tg[16];
#pragma unroll
    for (int i = 0; i < 16; ++i) { tp[i] = 0.f; tg[i] = 0.f; }
    for (int c = 0; c < Cc; ++c) {
      float xv = xp[(size_t)c * HW];
#pragma unroll
      for (int i = 0; i < 16; ++i) {
        tp[i] += wsp[(og * 16 + i) * Cc + c] * xv;
        tg[i] += wsg[(og * 16 + i) * Cc + c] * xv;
      }
    }
#pragma unroll
    for (int i = 0; i < 16; ++i) {
      accP[i] = fmaxf(accP[i], tp[i]);
      accG[i] = fmaxf(accG[i], tg[i]);
    }
  }
  bf16* pd = phi + ((size_t)b * Mp + m) * CI + og * 16;
  bf16x8 v0, v1;
#pragma unroll
  for (int i = 0; i < 8; ++i) {
    v0[i] = (bf16)(accP[i] + bphi[og * 16 + i]);
    v1[i] = (bf16)(accP[8 + i] + bphi[og * 16 + 8 + i]);
  }
  *(bf16x8*)pd = v0;
  *(bf16x8*)(pd + 8) = v1;
  bf16* gd = gT + (size_t)b * CI * Mp + (size_t)(og * 16) * Mp + m;
#pragma unroll
  for (int i = 0; i < 16; ++i)
    gd[(size_t)i * Mp] = (bf16)(accG[i] + bg[og * 16 + i]);
}

// ---------------- Kernel 3: streaming attention, one 16-row tile per wave ----------------
// y[b][n][c] = sum_m softmax_m(theta_n . phi_m) * g[c][m], f32 out.
// Shift-only softmax: p = exp(s - SM_SHIFT); the shift cancels between the
// P.G numerator and the ones-column denominator, so no running max / rescale
// and no cross-lane shuffles. Row sums come from an extra WMMA against an
// all-ones B matrix (each D column then holds the replicated row sum).

__global__ __launch_bounds__(256) void k_attn(const bf16* __restrict__ theta,
                                              const bf16* __restrict__ phi,
                                              const bf16* __restrict__ gT,
                                              float* __restrict__ y) {
  __shared__ bf16 pstage[8][16 * 32];  // per-wave P transpose staging (1 KB each)
  const int b = blockIdx.y;
  const int wave = threadIdx.x >> 5;
  const int lane = threadIdx.x & 31;
  const int r = lane & 15, h = lane >> 4;
  const int row0 = (blockIdx.x * 8 + wave) * 16;
  const bf16* thB = theta + (size_t)b * HW * CI;
  const bf16* phB = phi + (size_t)b * Mp * CI;
  const bf16* gB  = gT + (size_t)b * CI * Mp;
  float* yB = y + (size_t)b * HW * CI;
  bf16* ps = &pstage[wave][0];

  // Q tile fragments (reused across all 72 iterations)
  const bf16x16 a0 = load_a(thB + (size_t)row0 * CI, r, h, CI, 0);
  const bf16x16 a1 = load_a(thB + (size_t)row0 * CI, r, h, CI, 32);

  bf16x16 ones;
#pragma unroll
  for (int i = 0; i < 16; ++i) ones[i] = (bf16)1.0f;

  f32x8 o0 = {}, o1 = {}, o2 = {}, o3 = {}, lacc = {};
  const int rb = h * 8;

  for (int m0 = 0; m0 < Mp; m0 += 32) {
    // ---- S = theta_tile . phi_tile^T : two 16x16 tiles (32 key columns) ----
    f32x8 s0 = {}, s1 = {};
    {
      bf16x16 bb = load_b(phB + (size_t)m0 * CI, r, h, CI, 0);
      s0 = wmma_bf16(a0, bb, s0);
      bb = load_b(phB + (size_t)m0 * CI, r, h, CI, 32);
      s0 = wmma_bf16(a1, bb, s0);
      bb = load_b(phB + (size_t)(m0 + 16) * CI, r, h, CI, 0);
      s1 = wmma_bf16(a0, bb, s1);
      bb = load_b(phB + (size_t)(m0 + 16) * CI, r, h, CI, 32);
      s1 = wmma_bf16(a1, bb, s1);
    }
    // ---- shift-only exp (no cross-lane reductions) ----
    float p0[8], p1[8];
#pragma unroll
    for (int j = 0; j < 8; ++j) {
      p0[j] = __expf(s0[j] - SM_SHIFT);
      p1[j] = __expf(s1[j] - SM_SHIFT);
    }
    // ---- C-layout P -> LDS -> A-layout bf16 fragment (per-wave, in-order DS) ----
#pragma unroll
    for (int j = 0; j < 8; ++j) {
      ps[(rb + j) * 32 + r] = (bf16)p0[j];
      ps[(rb + j) * 32 + 16 + r] = (bf16)p1[j];
    }
    asm volatile("s_wait_dscnt 0x0" ::: "memory");
    bf16x16 pf = load_a(ps, r, h, 32, 0);
    // ---- O += P (16x32) . G (32x64); denominator via ones column ----
    o0 = wmma_bf16(pf, load_b(gB + (size_t)(0 * 16) * Mp + m0, r, h, Mp, 0), o0);
    o1 = wmma_bf16(pf, load_b(gB + (size_t)(1 * 16) * Mp + m0, r, h, Mp, 0), o1);
    o2 = wmma_bf16(pf, load_b(gB + (size_t)(2 * 16) * Mp + m0, r, h, Mp, 0), o2);
    o3 = wmma_bf16(pf, load_b(gB + (size_t)(3 * 16) * Mp + m0, r, h, Mp, 0), o3);
    lacc = wmma_bf16(pf, ones, lacc);
  }

  // ---- finalize: divide by softmax denominator, store y tile ----
#pragma unroll
  for (int j = 0; j < 8; ++j) {
    float rinv = 1.0f / lacc[j];
    size_t rowoff = (size_t)(row0 + rb + j) * CI;
    yB[rowoff + 0 * 16 + r] = o0[j] * rinv;
    yB[rowoff + 1 * 16 + r] = o1[j] * rinv;
    yB[rowoff + 2 * 16 + r] = o2[j] * rinv;
    yB[rowoff + 3 * 16 + r] = o3[j] * rinv;
  }
}

// ---------------- Kernel 4: z = conv1x1(y, w_W) + b_W -> f32 [B][128][N] ----------------

__global__ __launch_bounds__(256) void k_conv2(const float* __restrict__ y,
                                               const float* __restrict__ wW,
                                               const float* __restrict__ bW,
                                               float* __restrict__ z) {
  __shared__ float ws[Cc * CI];  // 32 KB
  const int b = blockIdx.y;
  const int n0 = blockIdx.x * 32;
  const int tid = threadIdx.x;
  for (int i = tid; i < Cc * CI; i += 256) ws[i] = wW[i];
  __syncthreads();
  const int pix = tid & 31, og = tid >> 5;  // og uniform per wave
  const int n = n0 + pix;
  const float* yr = y + ((size_t)b * HW + n) * CI;
  float acc[16];
#pragma unroll
  for (int i = 0; i < 16; ++i) acc[i] = 0.f;
  for (int c = 0; c < CI; ++c) {
    float yv = yr[c];
#pragma unroll
    for (int i = 0; i < 16; ++i) acc[i] += ws[(og * 16 + i) * CI + c] * yv;
  }
  float* zb = z + (size_t)b * Cc * HW;
#pragma unroll
  for (int i = 0; i < 16; ++i)
    zb[(size_t)(og * 16 + i) * HW + n] = acc[i] + bW[og * 16 + i];
}

// ---------------- Kernel 5: group-norm statistics ----------------

__global__ __launch_bounds__(256) void k_stats(const float* __restrict__ z,
                                               float* __restrict__ stats) {
  __shared__ float rs[256], rq[256];
  const int g = blockIdx.x, b = blockIdx.y;
  const float* zg = z + (size_t)b * Cc * HW + (size_t)(g * 4) * HW;
  float s = 0.f, q = 0.f;
  for (int i = threadIdx.x; i < 4 * HW; i += 256) {
    float v = zg[i];
    s += v; q += v * v;
  }
  rs[threadIdx.x] = s; rq[threadIdx.x] = q;
  __syncthreads();
  for (int st = 128; st > 0; st >>= 1) {
    if ((int)threadIdx.x < st) {
      rs[threadIdx.x] += rs[threadIdx.x + st];
      rq[threadIdx.x] += rq[threadIdx.x + st];
    }
    __syncthreads();
  }
  if (threadIdx.x == 0) {
    const float inv = 1.0f / (float)(4 * HW);
    float mean = rs[0] * inv;
    float var = rq[0] * inv - mean * mean;
    stats[(b * GN_GROUPS + g) * 2 + 0] = mean;
    stats[(b * GN_GROUPS + g) * 2 + 1] = rsqrtf(var + GN_EPS);
  }
}

// ---------------- Kernel 6: normalize + affine + residual ----------------

__global__ __launch_bounds__(256) void k_final(const float* __restrict__ z,
                                               const float* __restrict__ stats,
                                               const float* __restrict__ gnw,
                                               const float* __restrict__ gnb,
                                               const float* __restrict__ x,
                                               float* __restrict__ out) {
  size_t idx = (size_t)blockIdx.x * 256 + threadIdx.x;
  const size_t total = (size_t)Bn * Cc * HW;
  if (idx >= total) return;
  int c = (int)((idx / HW) % Cc);
  int b = (int)(idx / ((size_t)Cc * HW));
  int g = c >> 2;
  float mean = stats[(b * GN_GROUPS + g) * 2 + 0];
  float rstd = stats[(b * GN_GROUPS + g) * 2 + 1];
  out[idx] = (z[idx] - mean) * rstd * gnw[c] + gnb[c] + x[idx];
}

// ---------------- launch ----------------

extern "C" void kernel_launch(void* const* d_in, const int* in_sizes, int n_in,
                              void* d_out, int out_size, void* d_ws, size_t ws_size,
                              hipStream_t stream) {
  (void)in_sizes; (void)n_in; (void)out_size; (void)ws_size;
  const float* x     = (const float*)d_in[0];
  const float* w_th  = (const float*)d_in[1];
  const float* b_th  = (const float*)d_in[2];
  const float* w_phi = (const float*)d_in[3];
  const float* b_phi = (const float*)d_in[4];
  const float* w_g   = (const float*)d_in[5];
  const float* b_g   = (const float*)d_in[6];
  const float* w_W   = (const float*)d_in[7];
  const float* b_W   = (const float*)d_in[8];
  const float* gn_w  = (const float*)d_in[9];
  const float* gn_b  = (const float*)d_in[10];
  float* out = (float*)d_out;

  char* ws = (char*)d_ws;
  size_t off = 0;
  bf16* theta = (bf16*)(ws + off); off += (size_t)Bn * HW * CI * 2;  // 4.72 MB
  bf16* phi   = (bf16*)(ws + off); off += (size_t)Bn * Mp * CI * 2;  // 1.18 MB
  bf16* gT    = (bf16*)(ws + off); off += (size_t)Bn * CI * Mp * 2;  // 1.18 MB
  float* y    = (float*)(ws + off); off += (size_t)Bn * HW * CI * 4; // 9.44 MB
  float* z    = (float*)(ws + off); off += (size_t)Bn * Cc * HW * 4; // 18.9 MB
  float* stats = (float*)(ws + off); off += (size_t)Bn * GN_GROUPS * 2 * 4;

  k_theta<<<dim3(HW / 64, Bn), 256, 0, stream>>>(x, w_th, b_th, theta);
  k_phig<<<dim3(Mp / 64, Bn), 256, 0, stream>>>(x, w_phi, b_phi, w_g, b_g, phi, gT);
  k_attn<<<dim3(HW / 128, Bn), 256, 0, stream>>>(theta, phi, gT, y);
  k_conv2<<<dim3(HW / 32, Bn), 256, 0, stream>>>(y, w_W, b_W, z);
  k_stats<<<dim3(GN_GROUPS, Bn), 256, 0, stream>>>(z, stats);
  const size_t total = (size_t)Bn * Cc * HW;
  k_final<<<dim3((unsigned)((total + 255) / 256)), 256, 0, stream>>>(z, stats, gn_w, gn_b, x, out);
}
